// MultiheadAttention_68934225101453
// MI455X (gfx1250) — matrix operand order repaired and checked by reading, not compiled, not run on hardware
//
#include <hip/hip_runtime.h>

// Problem constants (B,T,C,H fixed by the reference)
#define BB 4
#define TT 2048
#define CC 1024
#define HH 16
#define HDIM 64
#define MTOK (BB * TT)          // 8192 tokens
#define SCALE_Q 0.125f          // 1/sqrt(64), folded into Q at QKV time

typedef __attribute__((ext_vector_type(16))) __bf16 v16bf;
typedef __attribute__((ext_vector_type(8)))  __bf16 v8bf;
typedef __attribute__((ext_vector_type(8)))  float  v8f;

// ---------------------------------------------------------------------------
// WMMA helper: D = A(16x32 bf16) * B(32x16 bf16) + C(16x16 f32)
// ---------------------------------------------------------------------------
static __device__ __forceinline__ v8f wmma_bf16(v16bf a, v16bf b, v8f c) {
  return __builtin_amdgcn_wmma_f32_16x16x32_bf16(
      /*neg_a=*/false, a, /*neg_b=*/false, b,
      /*c_mod=*/(short)0, c, /*reuse_a=*/false, /*reuse_b=*/false);
}

// A-fragment loader: A is row-major [*, lda] bf16. 16x32 tile at (row0, k0).
// Lane l: row = row0 + (l&15); K base = 16*(l>>4);
//   elements 0..15 = A[row][k0+base .. +15]
static __device__ __forceinline__ v16bf load_a_frag(const __bf16* A, int lda,
                                                    int row0, int k0) {
  const int lane = threadIdx.x & 31;
  const __bf16* p =
      A + (size_t)(row0 + (lane & 15)) * lda + k0 + ((lane >> 4) << 4);
  v8bf lo = *(const v8bf*)(p);
  v8bf hi = *(const v8bf*)(p + 8);
  v16bf f;
#pragma unroll
  for (int i = 0; i < 8; ++i) { f[i] = lo[i]; f[i + 8] = hi[i]; }
  return f;
}

// B-fragment loader where the logical B[k][n] is stored transposed as
// Bt[n][k] row-major with leading dim ldb (torch's [out,in] weight layout,
// a K row, and a vT row are all this shape). Lane l: col = col0 + (l&15);
// K = k0 + 16*(l>>4) + i  -> one contiguous 32-byte load.
static __device__ __forceinline__ v16bf load_b_frag_t(const __bf16* Bt, int ldb,
                                                      int col0, int k0) {
  const int lane = threadIdx.x & 31;
  const __bf16* p =
      Bt + (size_t)(col0 + (lane & 15)) * ldb + k0 + ((lane >> 4) << 4);
  return *(const v16bf*)(p);
}

// 16-lane (half-wave) reductions; rows of a WMMA C/D tile live within a
// 16-lane half, so xor masks 1..8 never cross halves.
static __device__ __forceinline__ float rmax16(float x) {
#pragma unroll
  for (int m = 1; m < 16; m <<= 1) x = fmaxf(x, __shfl_xor(x, m, 32));
  return x;
}
static __device__ __forceinline__ float rsum16(float x) {
#pragma unroll
  for (int m = 1; m < 16; m <<= 1) x += __shfl_xor(x, m, 32);
  return x;
}

// ---------------------------------------------------------------------------
// Kernel 1: fp32 -> bf16 conversion (4 elements / thread)
// ---------------------------------------------------------------------------
__global__ void cvt_bf16_kernel(const float* __restrict__ src,
                                __bf16* __restrict__ dst, int n) {
  int i = (blockIdx.x * blockDim.x + threadIdx.x) * 4;
  if (i < n) {
    float4 v = *(const float4*)(src + i);
    dst[i + 0] = (__bf16)v.x;
    dst[i + 1] = (__bf16)v.y;
    dst[i + 2] = (__bf16)v.z;
    dst[i + 3] = (__bf16)v.w;
  }
}

// ---------------------------------------------------------------------------
// Kernel 2: QKV projection  qkv = x @ W_attn^T + b_attn
//   X  : [MTOK, CC] bf16      Wa : [3C, C] bf16 (torch [out,in])
//   Q,K: [B,H,T,HD] bf16 (Q pre-scaled by 1/sqrt(HD))
//   VT : [B,H,HD,T] bf16 (transposed so PV B-fragments are contiguous)
// One wave -> 16x64 output strip; software-pipelined K loop, unrolled x2 so
// the register rotation coalesces into alternating register sets.
// grid (512, 6), 8 waves/block.
// ---------------------------------------------------------------------------
__global__ void __launch_bounds__(256) qkv_gemm_kernel(
    const __bf16* __restrict__ X, const __bf16* __restrict__ Wa,
    const float* __restrict__ ba, __bf16* __restrict__ Q,
    __bf16* __restrict__ K, __bf16* __restrict__ VT) {
  const int wave = threadIdx.x >> 5;
  const int lane = threadIdx.x & 31;
  const int m0 = blockIdx.x * 16;
  const int n0 = (blockIdx.y * 8 + wave) * 64;   // 64-wide strip

  v8f acc0 = {}, acc1 = {}, acc2 = {}, acc3 = {};

  // pipeline prologue
  v16bf a  = load_a_frag(X, CC, m0, 0);
  v16bf b0 = load_b_frag_t(Wa, CC, n0 +  0, 0);
  v16bf b1 = load_b_frag_t(Wa, CC, n0 + 16, 0);
  v16bf b2 = load_b_frag_t(Wa, CC, n0 + 32, 0);
  v16bf b3 = load_b_frag_t(Wa, CC, n0 + 48, 0);

#pragma unroll 2
  for (int k0 = 0; k0 < CC - 32; k0 += 32) {
    const int kn = k0 + 32;
    // prefetch the A row one more step ahead into the near cache
    __builtin_prefetch(X + (size_t)(m0 + (lane & 15)) * CC + kn + 32, 0, 3);
    // issue next step's loads first (fresh registers -> overlap with WMMAs)
    v16bf an  = load_a_frag(X, CC, m0, kn);
    v16bf bn0 = load_b_frag_t(Wa, CC, n0 +  0, kn);
    v16bf bn1 = load_b_frag_t(Wa, CC, n0 + 16, kn);
    v16bf bn2 = load_b_frag_t(Wa, CC, n0 + 32, kn);
    v16bf bn3 = load_b_frag_t(Wa, CC, n0 + 48, kn);
    acc0 = wmma_bf16(a, b0, acc0);
    acc1 = wmma_bf16(a, b1, acc1);
    acc2 = wmma_bf16(a, b2, acc2);
    acc3 = wmma_bf16(a, b3, acc3);
    a = an; b0 = bn0; b1 = bn1; b2 = bn2; b3 = bn3;
  }
  // drain
  acc0 = wmma_bf16(a, b0, acc0);
  acc1 = wmma_bf16(a, b1, acc1);
  acc2 = wmma_bf16(a, b2, acc2);
  acc3 = wmma_bf16(a, b3, acc3);

  const int rbase = m0 + ((lane >> 4) << 3);
#define QKV_EPILOG(ACC, TOFF)                                                  \
  {                                                                            \
    const int n = n0 + (TOFF) + (lane & 15);                                   \
    const float bias = ba[n];                                                  \
    const int sect = n / CC;                                                   \
    const int cc = n % CC;                                                     \
    const int h = cc / HDIM, d = cc % HDIM;                                    \
    _Pragma("unroll") for (int j = 0; j < 8; ++j) {                            \
      const int m = rbase + j;                                                 \
      const int b_ = m / TT, tt = m % TT;                                      \
      const float v = (ACC)[j] + bias;                                         \
      if (sect == 0) {                                                         \
        Q[(((size_t)(b_ * HH + h)) * TT + tt) * HDIM + d] =                    \
            (__bf16)(v * SCALE_Q);                                             \
      } else if (sect == 1) {                                                  \
        K[(((size_t)(b_ * HH + h)) * TT + tt) * HDIM + d] = (__bf16)v;         \
      } else {                                                                 \
        VT[(((size_t)(b_ * HH + h)) * HDIM + d) * TT + tt] = (__bf16)v;        \
      }                                                                        \
    }                                                                          \
  }
  QKV_EPILOG(acc0, 0)
  QKV_EPILOG(acc1, 16)
  QKV_EPILOG(acc2, 32)
  QKV_EPILOG(acc3, 48)
#undef QKV_EPILOG
}

// ---------------------------------------------------------------------------
// Kernel 3: flash attention forward. One wave owns a 16-row q tile and
// streams over keys 32 at a time. S never touches global memory.
// grid (B*H=64, 16), 8 waves/block, 8 KB LDS (1 KB P tile per wave).
// ---------------------------------------------------------------------------
__global__ void __launch_bounds__(256) attn_kernel(
    const __bf16* __restrict__ Q, const __bf16* __restrict__ K,
    const __bf16* __restrict__ VT, __bf16* __restrict__ AO) {
  __shared__ __align__(16) __bf16 plds_all[8][16 * 32];
  const int wave = threadIdx.x >> 5;
  const int lane = threadIdx.x & 31;
  __bf16* plds = &plds_all[wave][0];

  const int bh = blockIdx.x;                 // 0..63
  const int qtile = blockIdx.y * 8 + wave;   // 0..127
  const int b_ = bh / HH, h = bh % HH;
  const __bf16* Qh = Q + (size_t)bh * TT * HDIM;
  const __bf16* Kh = K + (size_t)bh * TT * HDIM;
  const __bf16* Vh = VT + (size_t)bh * HDIM * TT;
  const int q0 = qtile * 16;

  // Q fragments for the whole tile (K-dim = HD = 64 -> two 32-chunks)
  const v16bf qa0 = load_a_frag(Qh, HDIM, q0, 0);
  const v16bf qa1 = load_a_frag(Qh, HDIM, q0, 32);

  v8f o0 = {}, o1 = {}, o2 = {}, o3 = {};
  float mrow[8], lrow[8];
#pragma unroll
  for (int j = 0; j < 8; ++j) { mrow[j] = -3.0e38f; lrow[j] = 0.0f; }

  for (int kt = 0; kt < TT; kt += 32) {
    // ---- S = (Q/sqrt(d)) K^T : two 16x16 tiles covering 32 keys ----
    v16bf kb00 = load_b_frag_t(Kh, HDIM, kt, 0);
    v16bf kb01 = load_b_frag_t(Kh, HDIM, kt, 32);
    v16bf kb10 = load_b_frag_t(Kh, HDIM, kt + 16, 0);
    v16bf kb11 = load_b_frag_t(Kh, HDIM, kt + 16, 32);
    v8f s0 = {}, s1 = {};
    s0 = wmma_bf16(qa0, kb00, s0);
    s0 = wmma_bf16(qa1, kb01, s0);
    s1 = wmma_bf16(qa0, kb10, s1);
    s1 = wmma_bf16(qa1, kb11, s1);

    // V fragments can load while the softmax VALU work runs
    v16bf vb0 = load_b_frag_t(Vh, TT, 0, kt);
    v16bf vb1 = load_b_frag_t(Vh, TT, 16, kt);
    v16bf vb2 = load_b_frag_t(Vh, TT, 32, kt);
    v16bf vb3 = load_b_frag_t(Vh, TT, 48, kt);

    // ---- online softmax over these 32 columns ----
#pragma unroll
    for (int j = 0; j < 8; ++j) {
      float cmax = rmax16(fmaxf(s0[j], s1[j]));
      float mnew = fmaxf(mrow[j], cmax);
      float corr = __expf(mrow[j] - mnew);
      float p0 = __expf(s0[j] - mnew);
      float p1 = __expf(s1[j] - mnew);
      lrow[j] = lrow[j] * corr + rsum16(p0 + p1);
      mrow[j] = mnew;
      o0[j] *= corr; o1[j] *= corr; o2[j] *= corr; o3[j] *= corr;
      // stash P (D-layout) into LDS as a row-major 16x32 bf16 tile
      const int r = j + ((lane >> 4) << 3);
      plds[r * 32 + (lane & 15)] = (__bf16)p0;
      plds[r * 32 + 16 + (lane & 15)] = (__bf16)p1;
    }
    // same-wave DS ordering + keep compiler from hoisting the reads
    asm volatile("s_wait_dscnt 0" ::: "memory");

    // ---- O += P @ V  (A = P from LDS transpose-trip, B rows from vT) ----
    const v16bf pa = load_a_frag(plds, 32, 0, 0);
    o0 = wmma_bf16(pa, vb0, o0);
    o1 = wmma_bf16(pa, vb1, o1);
    o2 = wmma_bf16(pa, vb2, o2);
    o3 = wmma_bf16(pa, vb3, o3);
  }

  // ---- finalize: divide by row sums, store token-major bf16 [MTOK, CC] ----
  const int col = lane & 15;
#pragma unroll
  for (int j = 0; j < 8; ++j) {
    const float inv = 1.0f / lrow[j];
    const int t = q0 + j + ((lane >> 4) << 3);
    const size_t base = ((size_t)(b_ * TT) + t) * CC + h * HDIM;
    AO[base + 0 + col]  = (__bf16)(o0[j] * inv);
    AO[base + 16 + col] = (__bf16)(o1[j] * inv);
    AO[base + 32 + col] = (__bf16)(o2[j] * inv);
    AO[base + 48 + col] = (__bf16)(o3[j] * inv);
  }
}

// ---------------------------------------------------------------------------
// Kernel 4: output projection  out = attn_out @ W_proj^T + b_proj  (fp32 out)
// One wave -> 16x64 output strip, software-pipelined + unroll x2.
// grid (512, 2), 8 waves/block.
// ---------------------------------------------------------------------------
__global__ void __launch_bounds__(256) proj_gemm_kernel(
    const __bf16* __restrict__ AO, const __bf16* __restrict__ Wp,
    const float* __restrict__ bp, float* __restrict__ OUT) {
  const int wave = threadIdx.x >> 5;
  const int lane = threadIdx.x & 31;
  const int m0 = blockIdx.x * 16;
  const int n0 = (blockIdx.y * 8 + wave) * 64;

  v8f acc0 = {}, acc1 = {}, acc2 = {}, acc3 = {};

  v16bf a  = load_a_frag(AO, CC, m0, 0);
  v16bf b0 = load_b_frag_t(Wp, CC, n0 +  0, 0);
  v16bf b1 = load_b_frag_t(Wp, CC, n0 + 16, 0);
  v16bf b2 = load_b_frag_t(Wp, CC, n0 + 32, 0);
  v16bf b3 = load_b_frag_t(Wp, CC, n0 + 48, 0);

#pragma unroll 2
  for (int k0 = 0; k0 < CC - 32; k0 += 32) {
    const int kn = k0 + 32;
    __builtin_prefetch(AO + (size_t)(m0 + (lane & 15)) * CC + kn + 32, 0, 3);
    v16bf an  = load_a_frag(AO, CC, m0, kn);
    v16bf bn0 = load_b_frag_t(Wp, CC, n0 +  0, kn);
    v16bf bn1 = load_b_frag_t(Wp, CC, n0 + 16, kn);
    v16bf bn2 = load_b_frag_t(Wp, CC, n0 + 32, kn);
    v16bf bn3 = load_b_frag_t(Wp, CC, n0 + 48, kn);
    acc0 = wmma_bf16(a, b0, acc0);
    acc1 = wmma_bf16(a, b1, acc1);
    acc2 = wmma_bf16(a, b2, acc2);
    acc3 = wmma_bf16(a, b3, acc3);
    a = an; b0 = bn0; b1 = bn1; b2 = bn2; b3 = bn3;
  }
  acc0 = wmma_bf16(a, b0, acc0);
  acc1 = wmma_bf16(a, b1, acc1);
  acc2 = wmma_bf16(a, b2, acc2);
  acc3 = wmma_bf16(a, b3, acc3);

  const int rbase = m0 + ((lane >> 4) << 3);
#define PROJ_EPILOG(ACC, TOFF)                                                 \
  {                                                                            \
    const int n = n0 + (TOFF) + (lane & 15);                                   \
    const float bias = bp[n];                                                  \
    _Pragma("unroll") for (int j = 0; j < 8; ++j) {                            \
      OUT[(size_t)(rbase + j) * CC + n] = (ACC)[j] + bias;                     \
    }                                                                          \
  }
  PROJ_EPILOG(acc0, 0)
  PROJ_EPILOG(acc1, 16)
  PROJ_EPILOG(acc2, 32)
  PROJ_EPILOG(acc3, 48)
#undef PROJ_EPILOG
}

// ---------------------------------------------------------------------------
// Launcher
// ---------------------------------------------------------------------------
extern "C" void kernel_launch(void* const* d_in, const int* in_sizes, int n_in,
                              void* d_out, int out_size, void* d_ws,
                              size_t ws_size, hipStream_t stream) {
  const float* x    = (const float*)d_in[0];  // [B,T,C]
  const float* Wa_f = (const float*)d_in[1];  // [3C,C]
  const float* ba   = (const float*)d_in[2];  // [3C]
  const float* Wp_f = (const float*)d_in[3];  // [C,C]
  const float* bp   = (const float*)d_in[4];  // [C]
  float* out = (float*)d_out;                 // [B,T,C] fp32

  // Workspace carve-up (total ~88 MB)
  char* ws = (char*)d_ws;
  __bf16* xb  = (__bf16*)ws; ws += (size_t)MTOK * CC * 2;        // 16 MB
  __bf16* Wab = (__bf16*)ws; ws += (size_t)3 * CC * CC * 2;      //  6 MB
  __bf16* Wpb = (__bf16*)ws; ws += (size_t)CC * CC * 2;          //  2 MB
  __bf16* Qb  = (__bf16*)ws; ws += (size_t)MTOK * CC * 2;        // 16 MB
  __bf16* Kb  = (__bf16*)ws; ws += (size_t)MTOK * CC * 2;        // 16 MB
  __bf16* VTb = (__bf16*)ws; ws += (size_t)MTOK * CC * 2;        // 16 MB
  __bf16* AOb = (__bf16*)ws; ws += (size_t)MTOK * CC * 2;        // 16 MB

  // 1) fp32 -> bf16
  {
    const int nx = MTOK * CC;           // 8,388,608
    const int nwa = 3 * CC * CC;        // 3,145,728
    const int nwp = CC * CC;            // 1,048,576
    cvt_bf16_kernel<<<nx / 1024, 256, 0, stream>>>(x, xb, nx);
    cvt_bf16_kernel<<<nwa / 1024, 256, 0, stream>>>(Wa_f, Wab, nwa);
    cvt_bf16_kernel<<<nwp / 1024, 256, 0, stream>>>(Wp_f, Wpb, nwp);
  }

  // 2) QKV projection (M=8192 -> 512 m-tiles; N=3072 -> 48 strips of 64)
  qkv_gemm_kernel<<<dim3(MTOK / 16, 48 / 8), 256, 0, stream>>>(
      xb, Wab, ba, Qb, Kb, VTb);

  // 3) flash attention (64 head-batches x 128 q-tiles)
  attn_kernel<<<dim3(BB * HH, (TT / 16) / 8), 256, 0, stream>>>(
      Qb, Kb, VTb, AOb);

  // 4) output projection (N=1024 -> 16 strips of 64)
  proj_gemm_kernel<<<dim3(MTOK / 16, 16 / 8), 256, 0, stream>>>(
      AOb, Wpb, bp, out);
}
